// ESN_2937757630794
// MI455X (gfx1250) — matrix operand order repaired and checked by reading, compiled, not run on hardware
//
#include <hip/hip_runtime.h>
#include <hip/hip_bf16.h>
#include <math.h>

#define R      4096   // reservoir
#define INP    256    // input size
#define TSTEPS 4096
#define OUTN   256

typedef __attribute__((ext_vector_type(16))) __bf16 v16bf;
typedef __attribute__((ext_vector_type(8)))  __bf16 v8bf;
typedef __attribute__((ext_vector_type(8)))  float  v8f;

// ---------- helpers ----------
__device__ __forceinline__ unsigned short f2bf(float f) {
  unsigned u = __builtin_bit_cast(unsigned, f);
  unsigned r = u + 0x7FFFu + ((u >> 16) & 1u);   // round-to-nearest-even
  return (unsigned short)(r >> 16);
}
__device__ __forceinline__ float bf2f(unsigned short b) {
  return __builtin_bit_cast(float, (unsigned)b << 16);
}

// ---------- f32 -> bf16 conversion ----------
__global__ void cvt_bf16_kernel(const float* __restrict__ src,
                                unsigned short* __restrict__ dst, int n) {
  int i = blockIdx.x * blockDim.x + threadIdx.x;
  int stride = gridDim.x * blockDim.x;
  for (; i < n; i += stride) dst[i] = f2bf(src[i]);
}

// ---------- zero h double-buffer + barrier words ----------
__global__ void init_kernel(unsigned short* __restrict__ hbuf,
                            unsigned* __restrict__ bar) {
  int i = blockIdx.x * blockDim.x + threadIdx.x;
  if (i < 2 * R) hbuf[i] = 0;
  if (i < 2) bar[i] = 0;
}

// ---------- xin = x @ W_in^T via WMMA bf16 ----------
// one wave per 16x16 output tile; K = 256 -> 8 WMMAs
__global__ void xin_gemm_kernel(const unsigned short* __restrict__ xbf,    // [T][INP]
                                const unsigned short* __restrict__ winbf,  // [R][INP]
                                float* __restrict__ xin) {                 // [T][R]
  const int lane   = threadIdx.x & 31;
  const int waveId = (blockIdx.x * blockDim.x + threadIdx.x) >> 5;
  const int tt = waveId >> 8;        // t-tile 0..255
  const int rt = waveId & 255;       // r-tile 0..255
  const int t0 = tt * 16, r0 = rt * 16;
  const int m = lane & 15, g = lane >> 4;

  const unsigned short* xrow = xbf   + (size_t)(t0 + m) * INP;
  const unsigned short* wrow = winbf + (size_t)(r0 + m) * INP;

  v8f acc = {};
#pragma unroll
  for (int k0 = 0; k0 < INP; k0 += 32) {
    // A = x tile (16x32): lane m = M row, two 16B chunks per ISA layout
    v8bf alo = *(const v8bf*)(xrow + k0 + 8 * g);
    v8bf ahi = *(const v8bf*)(xrow + k0 + 16 + 8 * g);
    v16bf a = __builtin_shufflevector(alo, ahi, 0,1,2,3,4,5,6,7,8,9,10,11,12,13,14,15);
    // B = W_in^T tile (32x16): B[k][n] = W_in[r0+n][k]; two 16B chunks per ISA layout
    v8bf blo = *(const v8bf*)(wrow + k0 + 8 * g);
    v8bf bhi = *(const v8bf*)(wrow + k0 + 16 + 8 * g);
    v16bf b = __builtin_shufflevector(blo, bhi, 0,1,2,3,4,5,6,7,8,9,10,11,12,13,14,15);
    acc = __builtin_amdgcn_wmma_f32_16x16x32_bf16(false, a, false, b,
                                                  (short)0, acc, false, false);
  }
  // C/D layout: lane holds N = m, rows M = j + 8*g for VGPR j
  float* outp = xin + (size_t)(t0 + 8 * g) * R + r0 + m;
#pragma unroll
  for (int j = 0; j < 8; ++j) outp[(size_t)j * R] = acc[j];
}

// ---------- software grid barrier ----------
__device__ __forceinline__ void grid_barrier(unsigned* cnt, unsigned* gen,
                                             unsigned nblocks) {
  __syncthreads();
  if (threadIdx.x == 0) {
    __threadfence();
    unsigned g = __hip_atomic_load(gen, __ATOMIC_ACQUIRE, __HIP_MEMORY_SCOPE_AGENT);
    unsigned prev = __hip_atomic_fetch_add(cnt, 1u, __ATOMIC_ACQ_REL, __HIP_MEMORY_SCOPE_AGENT);
    if (prev == nblocks - 1u) {
      __hip_atomic_store(cnt, 0u, __ATOMIC_RELAXED, __HIP_MEMORY_SCOPE_AGENT);
      __hip_atomic_fetch_add(gen, 1u, __ATOMIC_RELEASE, __HIP_MEMORY_SCOPE_AGENT);
    } else {
      while (__hip_atomic_load(gen, __ATOMIC_ACQUIRE, __HIP_MEMORY_SCOPE_AGENT) == g) {
        __builtin_amdgcn_s_sleep(1);
      }
    }
  }
  __syncthreads();
}

// ---------- persistent recurrence: h = tanh(xin_t + W @ h) ----------
// 64 blocks x 4 waves = 256 waves; wave owns 16 rows of W; W stays L2-resident (32MB bf16)
#define REC_BLOCKS 64
#define REC_THREADS 128

__global__ void esn_recur_kernel(const unsigned short* __restrict__ Wbf,  // [R][R] bf16
                                 const float* __restrict__ xin,           // [T][R]
                                 unsigned short* __restrict__ hbuf,       // [2][R] bf16
                                 unsigned* __restrict__ bar) {
  const int lane = threadIdx.x & 31;
  const int wave = threadIdx.x >> 5;
  const int tile = blockIdx.x * (REC_THREADS / 32) + wave;  // 0..255
  const int r0 = tile * 16;
  const int m = lane & 15;
  const int g = lane >> 4;
  const unsigned short* Wrow = Wbf + (size_t)(r0 + m) * R;

  for (int t = 0; t < TSTEPS; ++t) {
    const unsigned short* hb = hbuf + (t & 1) * R;
    unsigned short* hn = hbuf + ((t + 1) & 1) * R;

    v8f acc = {};
#pragma unroll 2
    for (int k0 = 0; k0 < R; k0 += 32) {
      // A = W[16 rows x 32 k]
      v8bf alo = *(const v8bf*)(Wrow + k0 + 8 * g);
      v8bf ahi = *(const v8bf*)(Wrow + k0 + 16 + 8 * g);
      v16bf a = __builtin_shufflevector(alo, ahi, 0,1,2,3,4,5,6,7,8,9,10,11,12,13,14,15);
      // B = h broadcast into all 16 columns (address independent of lane%16)
      v8bf blo = *(const v8bf*)(hb + k0 + 8 * g);
      v8bf bhi = *(const v8bf*)(hb + k0 + 16 + 8 * g);
      v16bf b = __builtin_shufflevector(blo, bhi, 0,1,2,3,4,5,6,7,8,9,10,11,12,13,14,15);
      acc = __builtin_amdgcn_wmma_f32_16x16x32_bf16(false, a, false, b,
                                                    (short)0, acc, false, false);
    }
    // column 0 of D lives in lanes 0 (rows 0-7) and 16 (rows 8-15)
    if (m == 0) {
      const float* xr = xin + (size_t)t * R + r0 + g * 8;
      unsigned short* hr = hn + r0 + g * 8;
#pragma unroll
      for (int j = 0; j < 8; ++j)
        hr[j] = f2bf(tanhf(acc[j] + xr[j]));
    }
    grid_barrier(bar, bar + 1, (unsigned)gridDim.x);
  }
}

// ---------- readout: out = W_out_w @ h + b ----------
__global__ void readout_kernel(const unsigned short* __restrict__ hfin,
                               const float* __restrict__ Wo,
                               const float* __restrict__ bo,
                               float* __restrict__ out) {
  int o = blockIdx.x * blockDim.x + threadIdx.x;
  if (o >= OUTN) return;
  const float* wr = Wo + (size_t)o * R;
  float s = bo[o];
#pragma unroll 4
  for (int k = 0; k < R; ++k) s += wr[k] * bf2f(hfin[k]);
  out[o] = s;
}

extern "C" void kernel_launch(void* const* d_in, const int* in_sizes, int n_in,
                              void* d_out, int out_size, void* d_ws, size_t ws_size,
                              hipStream_t stream) {
  const float* x    = (const float*)d_in[0];  // [T][INP]
  const float* W_in = (const float*)d_in[1];  // [R][INP]
  const float* W    = (const float*)d_in[2];  // [R][R]
  const float* Wo   = (const float*)d_in[3];  // [OUTN][R]
  const float* bo   = (const float*)d_in[4];  // [OUTN]
  float* out = (float*)d_out;

  char* ws = (char*)d_ws;
  size_t off = 0;
  unsigned short* Wbf   = (unsigned short*)(ws + off); off += (size_t)R * R * 2;       // 32 MB
  float*          xin   = (float*)(ws + off);          off += (size_t)TSTEPS * R * 4;  // 64 MB
  unsigned short* xbf   = (unsigned short*)(ws + off); off += (size_t)TSTEPS * INP * 2;
  unsigned short* winbf = (unsigned short*)(ws + off); off += (size_t)R * INP * 2;
  unsigned short* hbuf  = (unsigned short*)(ws + off); off += (size_t)2 * R * 2;
  off = (off + 255) & ~(size_t)255;
  unsigned*       bar   = (unsigned*)(ws + off);

  // precision downconversion (once per launch)
  cvt_bf16_kernel<<<1024, 256, 0, stream>>>(W, Wbf, R * R);
  cvt_bf16_kernel<<<256, 256, 0, stream>>>(x, xbf, TSTEPS * INP);
  cvt_bf16_kernel<<<256, 256, 0, stream>>>(W_in, winbf, R * INP);
  init_kernel<<<32, 256, 0, stream>>>(hbuf, bar);

  // xin = x @ W_in^T : 256x256 tiles, 1 wave per tile, 4 waves per block
  xin_gemm_kernel<<<(256 * 256) / 4, 128, 0, stream>>>(xbf, winbf, xin);

  // persistent recurrence over T steps
  esn_recur_kernel<<<REC_BLOCKS, REC_THREADS, 0, stream>>>(Wbf, xin, hbuf, bar);

  // final h is in buffer half (TSTEPS & 1) == 0
  readout_kernel<<<(OUTN + 127) / 128, 128, 0, stream>>>(hbuf, Wo, bo, out);
}
